// BiRNNLayer_40879498728895
// MI455X (gfx1250) — matrix operand order repaired and checked by reading, not compiled, not run on hardware
//
#include <hip/hip_runtime.h>
#include <math.h>

// ---------------------------------------------------------------------------
// Bidirectional peephole LSTM for MI455X (gfx1250).
//   B=32, T=1024, D_IN=512, H=768, gates packed (i, j, f, o).
// Phase 1: W -> bf16 transposed (WMMA B-operand layout), x -> bf16 [T,B,D].
// Phase 2: Zx = x @ Wx + bias, parallel bf16 WMMA GEMM (2 M-tiles x 4 N-tiles
//          per wave), stored tile-major in the 16x16 f32 C-matrix lane layout.
// Phase 3: persistent scan kernel; 2 teams (fw/bw) x 6 blocks x 8 waves.
//          Each wave owns a 16-unit slice: 8 WMMA acc tiles (4 gates x 2 batch
//          halves), fully-unrolled K-loop over H=768 reading h_prev (bf16
//          ping-pong in ws) and L2-resident W_h columns; c-state in VGPRs.
//          Per-step: prefetch next Zx tiles (global_prefetch) during GEMM,
//          fence only h stores before the grid barrier, emit out stores after.
// ---------------------------------------------------------------------------

typedef __bf16 bf16_t;
typedef __attribute__((ext_vector_type(16))) __bf16 v16bf;
typedef __attribute__((ext_vector_type(8)))  __bf16 v8bf;
typedef __attribute__((ext_vector_type(8)))  float  v8f;

union V16 { v16bf v; v8bf h[2]; };

#define T_STEPS 1024
#define BATCH   32
#define DIN     512
#define HU      768
#define G4H     3072   // 4*H
#define NT      192    // G4H / 16  (N tiles)
#define UT      48     // H / 16    (unit tiles per gate)
#define NBLK    6      // blocks per direction in scan kernel

__device__ __forceinline__ float sigmoidf_(float x) {
  return 1.0f / (1.0f + __expf(-x));
}
__device__ __forceinline__ float tanhf_(float x) {
  float t = __expf(-2.0f * fabsf(x));
  float r = (1.0f - t) / (1.0f + t);
  return copysignf(r, x);
}

// ---------------------------------------------------------------------------
// Phase 1a: W [1280, 3072] f32 -> WxT [dir][3072][512] bf16 (col-major wrt GEMM)
//                               + WhT [dir][3072][768] bf16
__global__ void prep_weights(const float* __restrict__ Wfw,
                             const float* __restrict__ Wbw,
                             bf16_t* __restrict__ WxT,
                             bf16_t* __restrict__ WhT) {
  size_t i = (size_t)blockIdx.x * blockDim.x + threadIdx.x;
  const size_t per = (size_t)(DIN + HU) * G4H;
  if (i >= 2 * per) return;
  int dir = (i >= per) ? 1 : 0;
  size_t rem = i - (size_t)dir * per;
  int rk = (int)(rem / G4H);
  int n  = (int)(rem % G4H);
  float w = (dir ? Wbw : Wfw)[rem];
  if (rk < DIN)
    WxT[((size_t)dir * G4H + n) * DIN + rk] = (bf16_t)w;
  else
    WhT[((size_t)dir * G4H + n) * HU + (rk - DIN)] = (bf16_t)w;
}

// Phase 1b: x [B,T,D] f32 -> Xbf [T,B,D] bf16 (row index r = t*32+b matches
// the Zx GEMM M dimension).
__global__ void prep_x(const float* __restrict__ x, bf16_t* __restrict__ Xbf) {
  size_t i = (size_t)blockIdx.x * blockDim.x + threadIdx.x;
  if (i >= (size_t)BATCH * T_STEPS * DIN) return;
  int k = (int)(i % DIN);
  int t = (int)((i / DIN) % T_STEPS);
  int b = (int)(i / ((size_t)DIN * T_STEPS));
  Xbf[((size_t)t * BATCH + b) * DIN + k] = (bf16_t)x[i];
}

// Phase 1c: zero h ping-pong buffers and barrier counters each call.
__global__ void init_state(bf16_t* __restrict__ hbf, unsigned* __restrict__ ctr) {
  size_t i = (size_t)blockIdx.x * blockDim.x + threadIdx.x;
  const size_t n = 2ull * 2 * BATCH * HU;
  if (i < n) hbf[i] = (bf16_t)0.0f;
  if (i < 2) ctr[i] = 0u;
}

// ---------------------------------------------------------------------------
// Phase 2: Zx[dir][t][ntile][mt][256] (f32, C-matrix lane layout) =
//          X @ Wx + bias.  One wave: 2 M-tiles (both batch halves of one t)
//          x 4 consecutive N-tiles -> each B-tile feeds 2 WMMAs.
__global__ void zx_gemm(const bf16_t* __restrict__ Xbf,
                        const bf16_t* __restrict__ WxT,
                        const float* __restrict__ bias_fw,
                        const float* __restrict__ bias_bw,
                        float* __restrict__ Zx) {
  const int wave = threadIdx.x >> 5;
  const int lane = threadIdx.x & 31;
  const size_t gid = (size_t)blockIdx.x * 8 + wave;   // [0, 2*1024*48)
  const int dir  = (int)(gid / (1024 * 48));
  const int rem  = (int)(gid % (1024 * 48));
  const int t    = rem / 48;
  const int ngrp = rem % 48;          // group of 4 N tiles
  const int row  = lane & 15;
  const int krow = lane >> 4;
  const int kb8  = krow * 8;
  const int kb16 = krow * 16;

  const bf16_t* A0 = Xbf + ((size_t)t * BATCH + row) * DIN;   // batches 0..15
  const bf16_t* A1 = A0 + (size_t)16 * DIN;                   // batches 16..31
  const float* bias = dir ? bias_bw : bias_fw;

  v8f acc[4][2];
#pragma unroll
  for (int j = 0; j < 4; ++j) {
    float bv = bias[(ngrp * 4 + j) * 16 + row];
#pragma unroll
    for (int v = 0; v < 8; ++v) { acc[j][0][v] = bv; acc[j][1][v] = bv; }
  }

  for (int k0 = 0; k0 < DIN; k0 += 32) {
    V16 a0, a1;
    a0.h[0] = *(const v8bf*)(A0 + k0 + kb8);
    a0.h[1] = *(const v8bf*)(A0 + k0 + kb8 + 16);
    a1.h[0] = *(const v8bf*)(A1 + k0 + kb8);
    a1.h[1] = *(const v8bf*)(A1 + k0 + kb8 + 16);
#pragma unroll
    for (int j = 0; j < 4; ++j) {
      const int n = (ngrp * 4 + j) * 16 + row;
      V16 bb;
      bb.h[0] = *(const v8bf*)(WxT + ((size_t)dir * G4H + n) * DIN + k0 + kb8);
      bb.h[1] = *(const v8bf*)(WxT + ((size_t)dir * G4H + n) * DIN + k0 + kb8 + 16);
      const v16bf b = bb.v;
      acc[j][0] = __builtin_amdgcn_wmma_f32_16x16x32_bf16(
                      false, a0.v, false, b, (short)0, acc[j][0], false, false);
      acc[j][1] = __builtin_amdgcn_wmma_f32_16x16x32_bf16(
                      false, a1.v, false, b, (short)0, acc[j][1], false, false);
    }
  }

#pragma unroll
  for (int j = 0; j < 4; ++j) {
#pragma unroll
    for (int m = 0; m < 2; ++m) {
      const int ntile = ngrp * 4 + j;
      const size_t base =
          ((((size_t)dir * T_STEPS + t) * NT + ntile) * 2 + m) * 256 + (size_t)lane * 8;
      *(v8f*)(Zx + base) = acc[j][m];
    }
  }
}

// ---------------------------------------------------------------------------
// Phase 3: persistent sequential scan. 12 blocks total: blocks [0,6) = fw team,
// [6,12) = bw team. Wave -> one 16-unit tile; computes i/j/f/o for batches 0..31.
__global__ void lstm_scan(const float* __restrict__ Zx,
                          const bf16_t* __restrict__ WhT,
                          bf16_t* __restrict__ hbf,
                          const float* __restrict__ peep_fw,
                          const float* __restrict__ peep_bw,
                          float* __restrict__ out,
                          unsigned* __restrict__ ctr) {
  const int blk  = blockIdx.x;
  const int dir  = blk / NBLK;
  const int lblk = blk % NBLK;
  const int wave = threadIdx.x >> 5;
  const int lane = threadIdx.x & 31;
  const int ut   = lblk * 8 + wave;     // unit tile 0..47
  const int u0   = ut * 16;
  const int col  = lane & 15;           // unit within tile / C-matrix N
  const int krow = lane >> 4;
  const int kb8  = krow * 8;
  const int kb16 = krow * 16;

  const float*  peep = dir ? peep_bw : peep_fw;
  bf16_t* hb = hbf + (size_t)dir * 2 * BATCH * HU;
  unsigned* myctr = ctr + dir;

  // time-invariant W_h column base pointers, one per gate (offsets become
  // instruction immediates after full K-loop unroll)
  const bf16_t* Whbase = WhT + (size_t)dir * G4H * HU;
  const bf16_t* Wg[4];
#pragma unroll
  for (int g = 0; g < 4; ++g)
    Wg[g] = Whbase + ((size_t)g * HU + u0 + col) * HU + kb8;

  const float wi = peep[0 * HU + u0 + col];
  const float wf = peep[1 * HU + u0 + col];
  const float wo = peep[2 * HU + u0 + col];

  // cell state lives entirely in registers (only this wave touches it)
  float creg[2][8];
#pragma unroll
  for (int m = 0; m < 2; ++m)
#pragma unroll
    for (int v = 0; v < 8; ++v) creg[m][v] = 0.0f;

  float hnreg[2][8];

  for (int ts = 0; ts < T_STEPS; ++ts) {
    const int t = dir ? (T_STEPS - 1 - ts) : ts;
    const bf16_t* hprev = hb + (size_t)(ts & 1) * BATCH * HU;
    bf16_t*       hnext = hb + (size_t)((ts & 1) ^ 1) * BATCH * HU;

    // init accumulators from precomputed x-projection (+bias)
    v8f acc[4][2];
#pragma unroll
    for (int g = 0; g < 4; ++g)
#pragma unroll
      for (int m = 0; m < 2; ++m) {
        const int ntile = g * UT + ut;
        const size_t base =
            ((((size_t)dir * T_STEPS + t) * NT + ntile) * 2 + m) * 256 + (size_t)lane * 8;
        acc[g][m] = *(const v8f*)(Zx + base);
      }

    // prefetch next step's Zx tiles (HBM) so the post-barrier acc init hits
    // cache; each lane touches its own 32B slice -> all lines of the tile
    if (ts + 1 < T_STEPS) {
      const int tn = dir ? (t - 1) : (t + 1);
#pragma unroll
      for (int g = 0; g < 4; ++g)
#pragma unroll
        for (int m = 0; m < 2; ++m) {
          const int ntile = g * UT + ut;
          const size_t base =
              ((((size_t)dir * T_STEPS + tn) * NT + ntile) * 2 + m) * 256 + (size_t)lane * 8;
          __builtin_prefetch(Zx + base, 0, 1);
        }
    }

    // recurrent GEMM: h_prev [32,768] @ Wh columns (L2-resident)
#pragma unroll
    for (int k0 = 0; k0 < HU; k0 += 32) {
      V16 a0, a1;
      const bf16_t* pa = hprev + (size_t)col * HU + k0 + kb8;
      a0.h[0] = *(const v8bf*)pa;
      a0.h[1] = *(const v8bf*)(pa + 16);
      const bf16_t* pa1 = pa + 16 * HU;
      a1.h[0] = *(const v8bf*)pa1;
      a1.h[1] = *(const v8bf*)(pa1 + 16);
#pragma unroll
      for (int g = 0; g < 4; ++g) {
        V16 bb;
        bb.h[0] = *(const v8bf*)(Wg[g] + k0);
        bb.h[1] = *(const v8bf*)(Wg[g] + k0 + 16);
        const v16bf b = bb.v;
        acc[g][0] = __builtin_amdgcn_wmma_f32_16x16x32_bf16(
                        false, a0.v, false, b, (short)0, acc[g][0], false, false);
        acc[g][1] = __builtin_amdgcn_wmma_f32_16x16x32_bf16(
                        false, a1.v, false, b, (short)0, acc[g][1], false, false);
      }
    }

    // gates + state update; C layout: element v of lane L -> M = 8*krow+v, N = col
#pragma unroll
    for (int m = 0; m < 2; ++m) {
#pragma unroll
      for (int v = 0; v < 8; ++v) {
        const int batch = m * 16 + krow * 8 + v;
        const float c_old = creg[m][v];
        const float ig = sigmoidf_(acc[0][m][v] + wi * c_old);
        const float fg = sigmoidf_(acc[2][m][v] + 1.0f + wf * c_old);  // forget_bias=1
        const float cn = fg * c_old + ig * tanhf_(acc[1][m][v]);
        const float og = sigmoidf_(acc[3][m][v] + wo * cn);
        const float hn = og * tanhf_(cn);
        creg[m][v] = cn;
        hnreg[m][v] = hn;
        // only h feeds the other blocks -> only these stores sit on the
        // fenced (pre-barrier) path
        hnext[(size_t)batch * HU + u0 + col] = (bf16_t)hn;
      }
    }

    // per-direction grid barrier (monotonic counter, agent scope)
    __threadfence();
    __syncthreads();
    if (threadIdx.x == 0) {
      __hip_atomic_fetch_add(myctr, 1u, __ATOMIC_ACQ_REL, __HIP_MEMORY_SCOPE_AGENT);
      const unsigned target = (unsigned)(NBLK * (ts + 1));
      while (__hip_atomic_load(myctr, __ATOMIC_ACQUIRE, __HIP_MEMORY_SCOPE_AGENT) < target) {
        __builtin_amdgcn_s_sleep(2);
      }
    }
    __syncthreads();
    __threadfence();

    // HBM output stores issued after the barrier: their completion latency
    // overlaps the next step's GEMM instead of the release fence
#pragma unroll
    for (int m = 0; m < 2; ++m) {
#pragma unroll
      for (int v = 0; v < 8; ++v) {
        const int batch = m * 16 + krow * 8 + v;
        out[((size_t)batch * T_STEPS + t) * (2 * HU) + (size_t)dir * HU + u0 + col] =
            hnreg[m][v];
      }
    }
  }
}

// ---------------------------------------------------------------------------
extern "C" void kernel_launch(void* const* d_in, const int* in_sizes, int n_in,
                              void* d_out, int out_size, void* d_ws, size_t ws_size,
                              hipStream_t stream) {
  (void)in_sizes; (void)n_in; (void)out_size; (void)ws_size;
  const float* x   = (const float*)d_in[0];
  const float* Wfw = (const float*)d_in[1];
  const float* bfw = (const float*)d_in[2];
  const float* pfw = (const float*)d_in[3];
  const float* Wbw = (const float*)d_in[4];
  const float* bbw = (const float*)d_in[5];
  const float* pbw = (const float*)d_in[6];
  float* out = (float*)d_out;

  char* ws = (char*)d_ws;
  size_t off = 0;
  auto take = [&](size_t bytes) -> char* {
    char* p = ws + off;
    off = (off + bytes + 255) & ~(size_t)255;
    return p;
  };
  bf16_t* WxT = (bf16_t*)take(2ull * G4H * DIN * sizeof(bf16_t));      //   6.3 MB
  bf16_t* WhT = (bf16_t*)take(2ull * G4H * HU * sizeof(bf16_t));       //   9.4 MB
  bf16_t* Xbf = (bf16_t*)take((size_t)T_STEPS * BATCH * DIN * 2);      //  33.6 MB
  float*  Zx  = (float*)take(2ull * T_STEPS * NT * 2 * 256 * 4);       // 805.3 MB
  bf16_t* hbf = (bf16_t*)take(2ull * 2 * BATCH * HU * 2);              // 196 KB
  unsigned* ctr = (unsigned*)take(256);

  {
    const size_t n = 2ull * (DIN + HU) * G4H;
    prep_weights<<<(unsigned)((n + 255) / 256), 256, 0, stream>>>(Wfw, Wbw, WxT, WhT);
  }
  {
    const size_t n = (size_t)BATCH * T_STEPS * DIN;
    prep_x<<<(unsigned)((n + 255) / 256), 256, 0, stream>>>(x, Xbf);
  }
  {
    const size_t n = 2ull * 2 * BATCH * HU;
    init_state<<<(unsigned)((n + 255) / 256), 256, 0, stream>>>(hbf, ctr);
  }
  {
    // 2 dirs * 1024 t * 48 N-groups = 98304 waves / 8 waves per block
    zx_gemm<<<12288, 256, 0, stream>>>(Xbf, WxT, bfw, bbw, Zx);
  }
  {
    lstm_scan<<<2 * NBLK, 256, 0, stream>>>(Zx, WhT, hbf, pfw, pbw, out, ctr);
  }
}
